// DefectAwareAttention_55405078119238
// MI455X (gfx1250) — matrix operand order repaired and checked by reading, not compiled, not run on hardware
//
#include <hip/hip_runtime.h>

// Problem constants (from reference): B=2, N=2048, D=1024, H=16, HD=64
#define B_SZ   2
#define N_SEQ  2048
#define D_DIM  1024
#define H_NUM  16
#define HD_DIM 64
#define SCALE_F 0.125f   // HD^-0.5 = 1/8

typedef __attribute__((ext_vector_type(16))) __bf16 v16bf;
typedef __attribute__((ext_vector_type(8)))  __bf16 v8bf;
typedef __attribute__((ext_vector_type(8)))  float  v8f;

__device__ __forceinline__ v8bf ld8(const __bf16* p) {
  return *(const v8bf*)p;                 // 16B -> ds_load_b128 / global_load_b128
}
__device__ __forceinline__ v16bf concat8(v8bf lo, v8bf hi) {
  return __builtin_shufflevector(lo, hi, 0,1,2,3,4,5,6,7,8,9,10,11,12,13,14,15);
}
// A-fragment (16x32 bf16, row-major source row `rp`, 32 elems):
//   a[0..15] = k half*16 + 0..15
__device__ __forceinline__ v16bf a_frag(const __bf16* rp, int half) {
  return concat8(ld8(rp + half * 16), ld8(rp + half * 16 + 8));
}
// B-fragment (32x16 bf16, column-major source col `cp`, 32 elems):
//   b[0..15] = k half*16 + 0..15
__device__ __forceinline__ v16bf b_frag(const __bf16* cp, int half) {
  return concat8(ld8(cp + half * 16), ld8(cp + half * 16 + 8));
}

// CDNA5 async global->LDS copy (16B per lane), tracked by ASYNCcnt.
// ISA 10.2: generic address of an LDS object carries the LDS byte address in
// its low 32 bits, which is what VDST must hold for the async-to-LDS ops.
__device__ __forceinline__ void async_ld16(void* lds, const void* gsrc) {
  unsigned lds_off = (unsigned)(size_t)lds;
  asm volatile("global_load_async_to_lds_b128 %0, %1, off"
               :: "v"(lds_off), "v"(gsrc)
               : "memory");
}
// Async loads retire in issue order: waiting ASYNCcnt<=4 guarantees the
// previous tile's 4 issues are complete while the newest 4 stay in flight.
__device__ __forceinline__ void wait_async4() {
  asm volatile("s_wait_asynccnt 0x4" ::: "memory");
}
__device__ __forceinline__ void wait_async0() {
  asm volatile("s_wait_asynccnt 0x0" ::: "memory");
}

// ---------------------------------------------------------------- convert
__global__ void cvt_bf16_kernel(const float* __restrict__ in,
                                __bf16* __restrict__ out, int n) {
  int i = blockIdx.x * blockDim.x + threadIdx.x;
  if (i < n) out[i] = (__bf16)in[i];
}

// ---------------------------------------------------------------- QKV proj
// out[n, d] = x[n, :] . w[d, :] + bias[d]
// Q,K written as [B,H,N,HD]; V written transposed as [B,H,HD,N].
// grid: (M/64, D/64, 3), block: 128 (4 waves, 16 rows per wave)
__global__ __launch_bounds__(128) void qkv_proj_kernel(
    const __bf16* __restrict__ xb,
    const __bf16* __restrict__ wq, const float* __restrict__ bq,
    const __bf16* __restrict__ wk, const float* __restrict__ bk,
    const __bf16* __restrict__ wv, const float* __restrict__ bv,
    __bf16* __restrict__ Qb, __bf16* __restrict__ Kb, __bf16* __restrict__ Vb) {
  __shared__ __align__(16) __bf16 sA[2][64][32];   // x tile, row-major [row][k]
  __shared__ __align__(16) __bf16 sB[2][64][32];   // weight tile, column-major [col][k]

  const int tid  = threadIdx.x;
  const int wave = tid >> 5;
  const int lane = tid & 31;
  const int half = lane >> 4;
  const int lc   = lane & 15;

  const int mbase = blockIdx.x * 64;
  const int cbase = blockIdx.y * 64;
  const int which = blockIdx.z;

  const __bf16* W    = (which == 0) ? wq : (which == 1) ? wk : wv;
  const float*  bias = (which == 0) ? bq : (which == 1) ? bk : bv;
  __bf16*       Out  = (which == 0) ? Qb : (which == 1) ? Kb : Vb;

  // staging chunk coordinates (two 16B chunks per thread per tile)
  const int sr = tid >> 2;          // 0..31
  const int sc = (tid & 3) * 8;     // 0,8,16,24

  v8f acc[4];
#pragma unroll
  for (int g = 0; g < 4; ++g) acc[g] = v8f{0.f,0.f,0.f,0.f,0.f,0.f,0.f,0.f};

  auto stage = [&](int kb, int bi) {
    async_ld16(&sA[bi][sr][sc],      &xb[(size_t)(mbase + sr) * D_DIM + kb + sc]);
    async_ld16(&sA[bi][sr + 32][sc], &xb[(size_t)(mbase + sr + 32) * D_DIM + kb + sc]);
    async_ld16(&sB[bi][sr][sc],      &W[(size_t)(cbase + sr) * D_DIM + kb + sc]);
    async_ld16(&sB[bi][sr + 32][sc], &W[(size_t)(cbase + sr + 32) * D_DIM + kb + sc]);
  };

  const int nsteps = D_DIM / 32;      // 32
  stage(0, 0);
#pragma clang loop unroll(disable)
  for (int i = 0; i < nsteps; ++i) {
    if (i + 1 < nsteps) {             // prefetch next tile into other buffer
      stage((i + 1) * 32, (i + 1) & 1);
      wait_async4();                  // tile i resident (in-order retire)
    } else {
      wait_async0();
    }
    __syncthreads();
    {
      const int bi = i & 1;
      v16bf a = a_frag(&sA[bi][wave * 16 + lc][0], half);
#pragma unroll
      for (int g = 0; g < 4; ++g) {
        v16bf bfr = b_frag(&sB[bi][g * 16 + lc][0], half);
        acc[g] = __builtin_amdgcn_wmma_f32_16x16x32_bf16(
            false, a, false, bfr, (short)0, acc[g], false, false);
      }
    }
    __syncthreads();                  // all waves done reading before overwrite
  }

  // epilogue: bias, bf16, scatter (uniform branch on output layout)
  if (which == 2) {  // V stored transposed: [B,H,HD,N]
#pragma unroll
    for (int g = 0; g < 4; ++g) {
      int col = cbase + g * 16 + lc;
      float bv_ = bias[col];
      int h = col >> 6, hd = col & 63;
#pragma unroll
      for (int r = 0; r < 8; ++r) {
        int R = mbase + wave * 16 + r + 8 * half;
        int b = R >> 11, n = R & (N_SEQ - 1);
        Out[(((size_t)b * H_NUM + h) * HD_DIM + hd) * N_SEQ + n] =
            (__bf16)(acc[g][r] + bv_);
      }
    }
  } else {           // Q,K: [B,H,N,HD]
#pragma unroll
    for (int g = 0; g < 4; ++g) {
      int col = cbase + g * 16 + lc;
      float bv_ = bias[col];
      int h = col >> 6, hd = col & 63;
#pragma unroll
      for (int r = 0; r < 8; ++r) {
        int R = mbase + wave * 16 + r + 8 * half;
        int b = R >> 11, n = R & (N_SEQ - 1);
        Out[(((size_t)b * H_NUM + h) * N_SEQ + n) * HD_DIM + hd] =
            (__bf16)(acc[g][r] + bv_);
      }
    }
  }
}

// ---------------------------------------------------------------- flash attn
// grid: (B*H, N/64), block 128 (4 waves x 16 query rows). Online softmax,
// double-buffered async K/V staging.
__global__ __launch_bounds__(128) void flash_attn_kernel(
    const __bf16* __restrict__ Qb, const __bf16* __restrict__ Kb,
    const __bf16* __restrict__ Vb, const float* __restrict__ dp,
    __bf16* __restrict__ AOb) {
  __shared__ __align__(16) __bf16 sK[2][32][64];     // [key][d]  (B-cols for QK^T)
  __shared__ __align__(16) __bf16 sVt[2][64][32];    // [d][key]  (B-cols for P.V)
  __shared__ __align__(16) __bf16 sP[4][16][32];     // per-wave P ([q][key])

  const int tid  = threadIdx.x;
  const int wave = tid >> 5;
  const int lane = tid & 31;
  const int half = lane >> 4;
  const int lc   = lane & 15;

  const int bh = blockIdx.x;          // b*H + h
  const int b  = bh >> 4;             // H = 16
  const int h  = bh & 15;
  const int q0 = blockIdx.y * 64 + wave * 16;

  const __bf16* Qp  = Qb + (size_t)bh * N_SEQ * HD_DIM;
  const __bf16* Kp  = Kb + (size_t)bh * N_SEQ * HD_DIM;
  const __bf16* Vtp = Vb + (size_t)bh * HD_DIM * N_SEQ;   // transposed [HD][N]

  // staging chunk coordinates
  const int kr = tid >> 3;            // 0..15  (sK row), +16 for chunk 2
  const int kc = (tid & 7) * 8;       // 0..56
  const int vr = tid >> 2;            // 0..31  (sVt row), +32 for chunk 2
  const int vc = (tid & 3) * 8;       // 0..24

  // Q fragments (d 0..31 and 32..63) loaded straight from global (b128 loads)
  const __bf16* qr = Qp + (size_t)(q0 + lc) * HD_DIM;
  v16bf aq0 = a_frag(qr, half);
  v16bf aq1 = a_frag(qr + 32, half);

  // per-query additive bias (broadcast over keys; kept for fidelity)
  float prior[8];
#pragma unroll
  for (int r = 0; r < 8; ++r)
    prior[r] = dp[(size_t)b * N_SEQ + q0 + r + 8 * half];

  float mrow[8], lrow[8];
#pragma unroll
  for (int r = 0; r < 8; ++r) { mrow[r] = -__builtin_inff(); lrow[r] = 0.f; }
  v8f o[4];
#pragma unroll
  for (int g = 0; g < 4; ++g) o[g] = v8f{0.f,0.f,0.f,0.f,0.f,0.f,0.f,0.f};

  auto stage = [&](int j, int bi) {
    async_ld16(&sK[bi][kr][kc],       &Kp[(size_t)(j + kr) * HD_DIM + kc]);
    async_ld16(&sK[bi][kr + 16][kc],  &Kp[(size_t)(j + kr + 16) * HD_DIM + kc]);
    async_ld16(&sVt[bi][vr][vc],      &Vtp[(size_t)vr * N_SEQ + j + vc]);
    async_ld16(&sVt[bi][vr + 32][vc], &Vtp[(size_t)(vr + 32) * N_SEQ + j + vc]);
  };

  const int nsteps = N_SEQ / 32;      // 64
  stage(0, 0);
#pragma clang loop unroll(disable)
  for (int i = 0; i < nsteps; ++i) {
    if (i + 1 < nsteps) {             // prefetch next key slab
      stage((i + 1) * 32, (i + 1) & 1);
      wait_async4();
    } else {
      wait_async0();
    }
    __syncthreads();

    const int bi = i & 1;
    // S = Q . K^T  (two 16x16 C tiles: key cols 0-15 and 16-31)
    v8f s0 = v8f{0.f,0.f,0.f,0.f,0.f,0.f,0.f,0.f};
    v8f s1 = v8f{0.f,0.f,0.f,0.f,0.f,0.f,0.f,0.f};
#pragma unroll
    for (int cg = 0; cg < 2; ++cg) {
      const __bf16* krow = &sK[bi][cg * 16 + lc][0];
#pragma unroll
      for (int ds = 0; ds < 2; ++ds) {
        v16bf bfr = concat8(ld8(krow + ds * 32 + half * 16),
                            ld8(krow + ds * 32 + half * 16 + 8));
        v8f& s = (cg == 0) ? s0 : s1;
        s = __builtin_amdgcn_wmma_f32_16x16x32_bf16(
            false, (ds == 0) ? aq0 : aq1, false, bfr, (short)0, s, false, false);
      }
    }

    // online softmax over this 32-key slab
    float p0[8], p1[8], alpha[8];
#pragma unroll
    for (int r = 0; r < 8; ++r) {
      float v0 = s0[r] * SCALE_F + prior[r];
      float v1 = s1[r] * SCALE_F + prior[r];
      float mx = fmaxf(v0, v1);
#pragma unroll
      for (int msk = 1; msk <= 8; msk <<= 1)
        mx = fmaxf(mx, __shfl_xor(mx, msk, 32));
      float mnew = fmaxf(mrow[r], mx);
      alpha[r] = __expf(mrow[r] - mnew);
      p0[r] = __expf(v0 - mnew);
      p1[r] = __expf(v1 - mnew);
      float rs = p0[r] + p1[r];
#pragma unroll
      for (int msk = 1; msk <= 8; msk <<= 1)
        rs += __shfl_xor(rs, msk, 32);
      lrow[r] = lrow[r] * alpha[r] + rs;
      mrow[r] = mnew;
    }
#pragma unroll
    for (int g = 0; g < 4; ++g)
#pragma unroll
      for (int r = 0; r < 8; ++r) o[g][r] *= alpha[r];

    // C-layout f32 P -> row-major bf16 P in per-wave LDS scratch
#pragma unroll
    for (int r = 0; r < 8; ++r) {
      sP[wave][r + 8 * half][lc]      = (__bf16)p0[r];
      sP[wave][r + 8 * half][16 + lc] = (__bf16)p1[r];
    }
    v16bf ap = a_frag(&sP[wave][lc][0], half);

    // O += P . V   (B-cols = sVt rows)
#pragma unroll
    for (int g = 0; g < 4; ++g) {
      v16bf bfr = b_frag(&sVt[bi][g * 16 + lc][0], half);
      o[g] = __builtin_amdgcn_wmma_f32_16x16x32_bf16(
          false, ap, false, bfr, (short)0, o[g], false, false);
    }
    __syncthreads();
  }

  // normalize and write bf16 [B,N,D]
#pragma unroll
  for (int r = 0; r < 8; ++r) {
    float inv = 1.f / lrow[r];
    int R = q0 + r + 8 * half;
#pragma unroll
    for (int g = 0; g < 4; ++g) {
      int col = h * HD_DIM + g * 16 + lc;
      AOb[((size_t)b * N_SEQ + R) * D_DIM + col] = (__bf16)(o[g][r] * inv);
    }
  }
}

// ---------------------------------------------------------------- out proj
// d_out[n, d] = ao[n, :] . wo[d, :] + bo[d]   (f32 output)
__global__ __launch_bounds__(128) void out_proj_kernel(
    const __bf16* __restrict__ aob, const __bf16* __restrict__ wo,
    const float* __restrict__ bo, float* __restrict__ out) {
  __shared__ __align__(16) __bf16 sA[2][64][32];
  __shared__ __align__(16) __bf16 sB[2][64][32];   // column-major B: [col][k]

  const int tid  = threadIdx.x;
  const int wave = tid >> 5;
  const int lane = tid & 31;
  const int half = lane >> 4;
  const int lc   = lane & 15;

  const int mbase = blockIdx.x * 64;
  const int cbase = blockIdx.y * 64;

  const int sr = tid >> 2;
  const int sc = (tid & 3) * 8;

  v8f acc[4];
#pragma unroll
  for (int g = 0; g < 4; ++g) acc[g] = v8f{0.f,0.f,0.f,0.f,0.f,0.f,0.f,0.f};

  auto stage = [&](int kb, int bi) {
    async_ld16(&sA[bi][sr][sc],      &aob[(size_t)(mbase + sr) * D_DIM + kb + sc]);
    async_ld16(&sA[bi][sr + 32][sc], &aob[(size_t)(mbase + sr + 32) * D_DIM + kb + sc]);
    async_ld16(&sB[bi][sr][sc],      &wo[(size_t)(cbase + sr) * D_DIM + kb + sc]);
    async_ld16(&sB[bi][sr + 32][sc], &wo[(size_t)(cbase + sr + 32) * D_DIM + kb + sc]);
  };

  const int nsteps = D_DIM / 32;
  stage(0, 0);
#pragma clang loop unroll(disable)
  for (int i = 0; i < nsteps; ++i) {
    if (i + 1 < nsteps) {
      stage((i + 1) * 32, (i + 1) & 1);
      wait_async4();
    } else {
      wait_async0();
    }
    __syncthreads();
    {
      const int bi = i & 1;
      v16bf a = a_frag(&sA[bi][wave * 16 + lc][0], half);
#pragma unroll
      for (int g = 0; g < 4; ++g) {
        v16bf bfr = b_frag(&sB[bi][g * 16 + lc][0], half);
        acc[g] = __builtin_amdgcn_wmma_f32_16x16x32_bf16(
            false, a, false, bfr, (short)0, acc[g], false, false);
      }
    }
    __syncthreads();
  }

#pragma unroll
  for (int g = 0; g < 4; ++g) {
    int col = cbase + g * 16 + lc;
    float bv_ = bo[col];
#pragma unroll
    for (int r = 0; r < 8; ++r) {
      int R = mbase + wave * 16 + r + 8 * half;
      out[(size_t)R * D_DIM + col] = acc[g][r] + bv_;
    }
  }
}

// ---------------------------------------------------------------- launch
extern "C" void kernel_launch(void* const* d_in, const int* in_sizes, int n_in,
                              void* d_out, int out_size, void* d_ws, size_t ws_size,
                              hipStream_t stream) {
  (void)in_sizes; (void)n_in; (void)out_size; (void)ws_size;
  const float* x  = (const float*)d_in[0];
  const float* dp = (const float*)d_in[1];
  const float* wq = (const float*)d_in[2];
  const float* bq = (const float*)d_in[3];
  const float* wk = (const float*)d_in[4];
  const float* bk = (const float*)d_in[5];
  const float* wv = (const float*)d_in[6];
  const float* bv = (const float*)d_in[7];
  const float* wo = (const float*)d_in[8];
  const float* bo = (const float*)d_in[9];
  float* out = (float*)d_out;

  const size_t XN = (size_t)B_SZ * N_SEQ * D_DIM;  // 4,194,304
  const size_t WN = (size_t)D_DIM * D_DIM;         // 1,048,576

  __bf16* xb  = (__bf16*)d_ws;
  __bf16* wqb = xb  + XN;
  __bf16* wkb = wqb + WN;
  __bf16* wvb = wkb + WN;
  __bf16* wob = wvb + WN;
  __bf16* Qb  = wob + WN;
  __bf16* Kb  = Qb  + XN;
  __bf16* Vb  = Kb  + XN;   // V stored transposed per (b,h): [B,H,HD,N]
  __bf16* AOb = Vb  + XN;

  // 1) f32 -> bf16
  cvt_bf16_kernel<<<(int)((XN + 255) / 256), 256, 0, stream>>>(x,  xb,  (int)XN);
  cvt_bf16_kernel<<<(int)((WN + 255) / 256), 256, 0, stream>>>(wq, wqb, (int)WN);
  cvt_bf16_kernel<<<(int)((WN + 255) / 256), 256, 0, stream>>>(wk, wkb, (int)WN);
  cvt_bf16_kernel<<<(int)((WN + 255) / 256), 256, 0, stream>>>(wv, wvb, (int)WN);
  cvt_bf16_kernel<<<(int)((WN + 255) / 256), 256, 0, stream>>>(wo, wob, (int)WN);

  // 2) fused QKV projections (V transposed on store)
  dim3 gq((B_SZ * N_SEQ) / 64, D_DIM / 64, 3);
  qkv_proj_kernel<<<gq, 128, 0, stream>>>(xb, wqb, bq, wkb, bk, wvb, bv, Qb, Kb, Vb);

  // 3) flash attention
  dim3 ga(B_SZ * H_NUM, N_SEQ / 64);
  flash_attn_kernel<<<ga, 128, 0, stream>>>(Qb, Kb, Vb, dp, AOb);

  // 4) output projection
  dim3 go((B_SZ * N_SEQ) / 64, D_DIM / 64);
  out_proj_kernel<<<go, 128, 0, stream>>>(AOb, wob, bo, out);
}
